// Level1GNNEncoder_19292993094408
// MI455X (gfx1250) — matrix operand order repaired and checked by reading, not compiled, not run on hardware
//
#include <hip/hip_runtime.h>
#include <hip/hip_bf16.h>
#include <stdint.h>

// ---------------------------------------------------------------------------
// GIN encoder for MI455X (gfx1250, wave32).
//   per layer: agg = h + scatter_sum(h[src]->dst); z = relu(agg@W1+b1);
//              h = relu(z@W2+b2)
// GEMMs use v_wmma_f32_16x16x32_bf16 (bf16 A/B, f32 accumulate).
// fp32->bf16 conversion via __builtin_convertvector -> native packed cvt.
// Scatter uses native global_atomic_add_f32 (relaxed, agent scope).
// ---------------------------------------------------------------------------

typedef __attribute__((ext_vector_type(16))) __bf16  v16bf;
typedef __attribute__((ext_vector_type(16))) float   v16f;
typedef __attribute__((ext_vector_type(16))) unsigned short v16us;
typedef __attribute__((ext_vector_type(8)))  float   v8f;

// ---------------------------------------------------------------------------
// Pack a 128x128 row-major fp32 weight matrix into bf16 WMMA B-fragments.
// Fragment (nt,kt): lane holds column n = nt*16 + (lane&15), 16 contiguous K
// values starting at kt*32 + (lane<16 ? 0 : 16).  Contiguous 32B per lane.
// 6 matrices, 16384 bf16 each.
// ---------------------------------------------------------------------------
__global__ void prep_weights(const float* W0, const float* W1, const float* W2,
                             const float* W3, const float* W4, const float* W5,
                             __bf16* out) {
    int t = blockIdx.x * blockDim.x + threadIdx.x;
    if (t >= 6 * 16384) return;
    int mat  = t >> 14;
    int i    = t & 16383;
    int e    = i & 15;
    int lane = (i >> 4) & 31;
    int tile = i >> 9;          // 0..31  (nt*4 + kt)
    int nt   = tile >> 2;
    int kt   = tile & 3;
    int k = kt * 32 + ((lane & 16) ? 16 : 0) + e;
    int n = nt * 16 + (lane & 15);
    const float* W = (mat == 0) ? W0 : (mat == 1) ? W1 : (mat == 2) ? W2
                   : (mat == 3) ? W3 : (mat == 4) ? W4 : W5;
    out[t] = (__bf16)W[k * 128 + n];            // native cvt, RNE
}

// ---------------------------------------------------------------------------
// agg := h   (vectorized copy, float4)
// ---------------------------------------------------------------------------
__global__ void copy_f4(float4* __restrict__ dst, const float4* __restrict__ src,
                        int n4) {
    int i = blockIdx.x * blockDim.x + threadIdx.x;
    if (i < n4) dst[i] = src[i];
}

// ---------------------------------------------------------------------------
// agg[dst] += h[src]   — one wave per edge; lane owns 4 contiguous floats of
// the 128-wide feature row (full 512B row per wave, coalesced gather).
// ---------------------------------------------------------------------------
__global__ void gin_scatter(float* __restrict__ agg, const float* __restrict__ h,
                            const int* __restrict__ ei, int nE) {
    int e = blockIdx.x * (blockDim.x >> 5) + (threadIdx.x >> 5);
    if (e >= nE) return;
    int lane = threadIdx.x & 31;
    int s = ei[e];
    int d = ei[nE + e];
    const float4 v = *(const float4*)(h + (size_t)s * 128 + lane * 4);
    float* p = agg + (size_t)d * 128 + lane * 4;
    __hip_atomic_fetch_add(p + 0, v.x, __ATOMIC_RELAXED, __HIP_MEMORY_SCOPE_AGENT);
    __hip_atomic_fetch_add(p + 1, v.y, __ATOMIC_RELAXED, __HIP_MEMORY_SCOPE_AGENT);
    __hip_atomic_fetch_add(p + 2, v.z, __ATOMIC_RELAXED, __HIP_MEMORY_SCOPE_AGENT);
    __hip_atomic_fetch_add(p + 3, v.w, __ATOMIC_RELAXED, __HIP_MEMORY_SCOPE_AGENT);
}

// ---------------------------------------------------------------------------
// out = relu(A @ W + bias), A:[16*nTiles x 128] fp32, W packed bf16 fragments.
// One wave per 16-row strip.  A-fragments (4 x K=32) hoisted & converted to
// bf16 once (packed cvt); 8 column tiles x 4 v_wmma_f32_16x16x32_bf16 each.
// Row-local => safe to call with out == A (in-place).
// ---------------------------------------------------------------------------
__global__ void gin_gemm_bias_relu(const float* A, const __bf16* Bw,
                                   const float* __restrict__ bias, float* out,
                                   int nTiles) {
    const int w    = threadIdx.x >> 5;
    const int lane = threadIdx.x & 31;
    const int tile = blockIdx.x * (blockDim.x >> 5) + w;
    if (tile >= nTiles) return;                 // wave-uniform: EXEC stays full

    const int mrow = (lane & 15);               // A row within strip (both halves)
    const int hi8  = (lane & 16) ? 8 : 0;       // K-half / output-row-half select
    const float* ar = A + ((size_t)tile * 16 + mrow) * 128;

    // Hoist + convert the four 16x32 A fragments (per-lane: K 0..15
    // (+2*hi8) — lane L and L+16 cover complementary halves of the same row).
    v16bf afrag[4];
#pragma unroll
    for (int kt = 0; kt < 4; ++kt) {
        const int kb = kt * 32 + (hi8 << 1);
        float4 p0 = *(const float4*)(ar + kb);
        float4 p1 = *(const float4*)(ar + kb + 4);
        float4 p2 = *(const float4*)(ar + kb + 8);
        float4 p3 = *(const float4*)(ar + kb + 12);
        v16f vf;
        vf[0]  = p0.x; vf[1]  = p0.y; vf[2]  = p0.z; vf[3]  = p0.w;
        vf[4]  = p1.x; vf[5]  = p1.y; vf[6]  = p1.z; vf[7]  = p1.w;
        vf[8]  = p2.x; vf[9]  = p2.y; vf[10] = p2.z; vf[11] = p2.w;
        vf[12] = p3.x; vf[13] = p3.y; vf[14] = p3.z; vf[15] = p3.w;
        afrag[kt] = __builtin_convertvector(vf, v16bf);   // v_cvt_pk_bf16_f32
    }

    const v16us* bp = (const v16us*)Bw;         // fragment = bp[(nt*4+kt)*32 + lane]

#pragma unroll
    for (int nt = 0; nt < 8; ++nt) {
        const float bs = bias[nt * 16 + mrow];  // col = nt*16 + (lane&15)
        v8f acc = {bs, bs, bs, bs, bs, bs, bs, bs};
#pragma unroll
        for (int kt = 0; kt < 4; ++kt) {
            v16bf bb = __builtin_bit_cast(v16bf, bp[(nt * 4 + kt) * 32 + lane]);
            acc = __builtin_amdgcn_wmma_f32_16x16x32_bf16(
                      false, afrag[kt], false, bb, (short)0, acc, false, false);
        }
        // D layout: VGPR j -> row tile*16 + hi8 + j, col nt*16 + (lane&15)
        float* orow = out + ((size_t)tile * 16 + hi8) * 128 + nt * 16 + mrow;
#pragma unroll
        for (int j = 0; j < 8; ++j)
            orow[(size_t)j * 128] = fmaxf(acc[j], 0.0f);
    }
}

// ---------------------------------------------------------------------------
extern "C" void kernel_launch(void* const* d_in, const int* in_sizes, int n_in,
                              void* d_out, int out_size, void* d_ws, size_t ws_size,
                              hipStream_t stream) {
    const float* x  = (const float*)d_in[0];
    const int*   ei = (const int*)d_in[1];

    const int nNodes = in_sizes[0] / 128;   // 50000 (divisible by 16)
    const int nEdges = in_sizes[1] / 2;     // 800000
    const int nTiles = nNodes / 16;         // 3125
    const int nElem  = nNodes * 128;
    const int n4     = nElem / 4;

    float* B0 = (float*)d_ws;                                      // N x 128 scratch
    __bf16* wp = (__bf16*)
        (((uintptr_t)(B0 + nElem) + 255) & ~(uintptr_t)255);       // packed weights
    float* outF = (float*)d_out;

    // Pack all 6 weight matrices: [W1_0, W2_0, W1_1, W2_1, W1_2, W2_2]
    prep_weights<<<(6 * 16384 + 255) / 256, 256, 0, stream>>>(
        (const float*)d_in[2],  (const float*)d_in[4],
        (const float*)d_in[6],  (const float*)d_in[8],
        (const float*)d_in[10], (const float*)d_in[12], wp);

    const int copyBlocks    = (n4 + 255) / 256;
    const int scatterBlocks = (nEdges + 7) / 8;      // 8 waves (edges) per block
    const int gemmBlocks    = (nTiles + 7) / 8;      // 8 waves (strips) per block

    const float* hin = x;
    for (int l = 0; l < 3; ++l) {
        float* agg = (l == 1) ? outF : B0;           // rotate through d_out
        const float* b1 = (const float*)d_in[3 + 4 * l];
        const float* b2 = (const float*)d_in[5 + 4 * l];
        const __bf16* W1p = wp + (size_t)(2 * l)     * 16384;
        const __bf16* W2p = wp + (size_t)(2 * l + 1) * 16384;

        copy_f4<<<copyBlocks, 256, 0, stream>>>((float4*)agg, (const float4*)hin, n4);
        gin_scatter<<<scatterBlocks, 256, 0, stream>>>(agg, hin, ei, nEdges);
        // z = relu(agg @ W1 + b1)   (in-place: row-local)
        gin_gemm_bias_relu<<<gemmBlocks, 256, 0, stream>>>(agg, W1p, b1, agg, nTiles);
        // h = relu(z @ W2 + b2); final layer lands in d_out
        float* ho = (l == 2) ? outF : agg;
        gin_gemm_bias_relu<<<gemmBlocks, 256, 0, stream>>>(agg, W2p, b2, ho, nTiles);
        hin = ho;
    }
}
